// DlinearModel_50654844289789
// MI455X (gfx1250) — matrix operand order repaired and checked by reading, not compiled, not run on hardware
//
#include <hip/hip_runtime.h>

// ---------------------------------------------------------------------------
// DLinear on MI455X (gfx1250): single fused GEMM after folding the moving-
// average trend operator into an effective weight matrix, computed with
// bf16x3 split-precision V_WMMA_F32_16X16X32_BF16 (f32-quality results at
// bf16 matrix-core rates), async global->LDS staging, double buffering.
// ---------------------------------------------------------------------------

typedef __attribute__((ext_vector_type(16))) __bf16 v16bf;
typedef __attribute__((ext_vector_type(8)))  __bf16 v8bf;
typedef __attribute__((ext_vector_type(8)))  float  v8f;
typedef __attribute__((ext_vector_type(4)))  int    v4i;

#define T_LEN 336
#define C_LEN 1782
#define P_LEN 720
#define PW    768            // padded rows of W_eff (multiple of 64)
#define KP    352            // padded K (multiple of 32)
#define KS    32             // K per WMMA step
#define NK    (KP / KS)      // 11 K-steps
#define TN    64             // N tile (c) per block
#define TM    64             // M tile (p) per block: 4 waves x 16
#define SA    40             // LDS row stride in bf16 (80B, 16B-aligned rows)

#if defined(__has_builtin)
#  if __has_builtin(__builtin_amdgcn_global_load_async_to_lds_b128) && \
      __has_builtin(__builtin_amdgcn_s_wait_asynccnt)
#    define USE_ASYNC 1
#  endif
#endif
#ifndef USE_ASYNC
#  define USE_ASYNC 0
#endif

#if USE_ASYNC
// Builtin prototype (from hipcc diagnostic): (v4i addrspace(1)*, v4i addrspace(3)*, Imm, Imm)
typedef __attribute__((address_space(1))) v4i* g_v4i_p;
typedef __attribute__((address_space(3))) v4i* l_v4i_p;
#endif

// ---------------------------------------------------------------------------
// Prep: W_eff[p,s] = Ws[p,s] + 0.2 * sum_t mult(t,s) * (Wt-Ws)[p,t]
// (moving-average adjoint with replicate-pad multiplicities), pre-split into
// bf16 hi/lo, zero-padded to PW x KP so the GEMM needs no K/M edge masking.
// ---------------------------------------------------------------------------
__global__ __launch_bounds__(256) void prep_weights(
    const float* __restrict__ Ws, const float* __restrict__ Wt,
    __bf16* __restrict__ Whi, __bf16* __restrict__ Wlo) {
  int idx = blockIdx.x * blockDim.x + threadIdx.x;
  if (idx >= PW * KP) return;
  int p = idx / KP, s = idx % KP;
  float v = 0.0f;
  if (p < P_LEN && s < T_LEN) {
    const float* ws = Ws + (size_t)p * T_LEN;
    const float* wt = Wt + (size_t)p * T_LEN;
    float tsum = 0.0f;
    #pragma unroll
    for (int j = -2; j <= 2; ++j) {
      int t = s + j;
      if (t >= 0 && t < T_LEN) tsum += wt[t] - ws[t];
    }
    if (s == 0)
      tsum += 2.0f * (wt[0] - ws[0]) + (wt[1] - ws[1]);
    if (s == T_LEN - 1)
      tsum += 2.0f * (wt[T_LEN - 1] - ws[T_LEN - 1]) + (wt[T_LEN - 2] - ws[T_LEN - 2]);
    v = ws[s] + 0.2f * tsum;
  }
  __bf16 hi = (__bf16)v;
  __bf16 lo = (__bf16)(v - (float)hi);
  Whi[idx] = hi;
  Wlo[idx] = lo;
}

// ---------------------------------------------------------------------------
// GEMM: out[b,p,c] = relu( sum_t x[b,t,c] * W_eff[p,t] + bs[p] + bt[p] )
// Block = 128 threads (4 waves). Wave w computes M rows [p0+16w, p0+16w+16),
// N = 64 columns as 4 WMMA subtiles, K looped 32 at a time, double buffered.
// ---------------------------------------------------------------------------
__global__ __launch_bounds__(128) void dlinear_gemm(
    const float* __restrict__ x,
    const __bf16* __restrict__ Whi, const __bf16* __restrict__ Wlo,
    const float* __restrict__ bs, const float* __restrict__ bt,
    float* __restrict__ out) {
  __shared__ __align__(16) __bf16 sAhi[2][TM * SA];   // [p][k] bf16 hi
  __shared__ __align__(16) __bf16 sAlo[2][TM * SA];   // [p][k] bf16 lo
  __shared__ __align__(16) __bf16 sBhi[2][TN * SA];   // transposed: [n][k]
  __shared__ __align__(16) __bf16 sBlo[2][TN * SA];

  const int tid  = threadIdx.x;
  const int lane = tid & 31;
  const int wave = tid >> 5;
  const int half = lane >> 4;     // which 16-lane half of the wave
  const int l16  = lane & 15;

  const int c0 = blockIdx.x * TN;
  const int p0 = blockIdx.y * TM;
  const int b  = blockIdx.z;

  const float* xb = x + (size_t)b * T_LEN * C_LEN;

  v8f acc[4] = {};       // 4 N-subtiles of the 16x64 per-wave output strip
  float breg[16];        // x tile staged through registers (hi/lo split here)

  auto loadB = [&](int ks) {
    #pragma unroll
    for (int i = 0; i < 16; ++i) {
      int q = tid + 128 * i;                 // 32x64 f32 tile element
      int k = q >> 6;                        // 0..31
      int n = q & 63;                        // 0..63
      int kg = ks * KS + k; if (kg > T_LEN - 1) kg = T_LEN - 1;  // clamp: padded
      int cg = c0 + n;      if (cg > C_LEN - 1) cg = C_LEN - 1;  // A cols are 0
      breg[i] = xb[(size_t)kg * C_LEN + cg];
    }
  };
  auto storeB = [&](int bf) {
    #pragma unroll
    for (int i = 0; i < 16; ++i) {
      int q = tid + 128 * i;
      int k = q >> 6;
      int n = q & 63;
      float v = breg[i];
      __bf16 hi = (__bf16)v;
      __bf16 lo = (__bf16)(v - (float)hi);
      sBhi[bf][n * SA + k] = hi;             // transposed store [n][k]
      sBlo[bf][n * SA + k] = lo;
    }
  };
  auto issueA = [&](int bf, int ks) {
    #pragma unroll
    for (int i = 0; i < 4; ++i) {
      int cid = tid + 128 * i;               // 512 chunks of 8 bf16 (16B)
      int sel = cid >> 8;                    // 0 = hi tile, 1 = lo tile
      int r   = (cid & 255) >> 2;            // row 0..63
      int j   = cid & 3;                     // 8-bf16 chunk within 32-col row
      const __bf16* src = (sel ? Wlo : Whi) + (size_t)(p0 + r) * KP + ks * KS + j * 8;
      __bf16* dst = (sel ? sAlo[bf] : sAhi[bf]) + r * SA + j * 8;
#if USE_ASYNC
      v4i* sp = (v4i*)src;                   // drop const, retype (generic AS)
      v4i* dp = (v4i*)dst;
      __builtin_amdgcn_global_load_async_to_lds_b128((g_v4i_p)sp, (l_v4i_p)dp, 0, 0);
#else
      *(v8bf*)dst = *(const v8bf*)src;
#endif
    }
  };
  auto compute = [&](int bf) {
    // A fragment: lane half selects K-block per the 16-bit A layout:
    // element e -> K = e + (e<8 ? 0 : 8) + 8*half
    const int m = wave * 16 + l16;
    const __bf16* pAh = &sAhi[bf][m * SA + 8 * half];
    const __bf16* pAl = &sAlo[bf][m * SA + 8 * half];
    v8bf a0h = *(const v8bf*)(pAh);
    v8bf a1h = *(const v8bf*)(pAh + 16);
    v8bf a0l = *(const v8bf*)(pAl);
    v8bf a1l = *(const v8bf*)(pAl + 16);
    v16bf ahi = __builtin_shufflevector(a0h, a1h, 0,1,2,3,4,5,6,7,8,9,10,11,12,13,14,15);
    v16bf alo = __builtin_shufflevector(a0l, a1l, 0,1,2,3,4,5,6,7,8,9,10,11,12,13,14,15);
    #pragma unroll
    for (int s = 0; s < 4; ++s) {
      // B fragment: element e -> K = e + (e<8 ? 0 : 8) + 8*half, column N = l16 (from [n][k])
      const int n = s * 16 + l16;
      const __bf16* pBh = &sBhi[bf][n * SA + 8 * half];
      const __bf16* pBl = &sBlo[bf][n * SA + 8 * half];
      v8bf b0h = *(const v8bf*)(pBh);
      v8bf b1h = *(const v8bf*)(pBh + 16);
      v8bf b0l = *(const v8bf*)(pBl);
      v8bf b1l = *(const v8bf*)(pBl + 16);
      v16bf bhi = __builtin_shufflevector(b0h, b1h, 0,1,2,3,4,5,6,7,8,9,10,11,12,13,14,15);
      v16bf blo = __builtin_shufflevector(b0l, b1l, 0,1,2,3,4,5,6,7,8,9,10,11,12,13,14,15);
      // bf16x3: hi*hi + hi*lo + lo*hi, all accumulated in f32
      acc[s] = __builtin_amdgcn_wmma_f32_16x16x32_bf16(
          false, ahi, false, bhi, (short)0, acc[s], false, false);
      acc[s] = __builtin_amdgcn_wmma_f32_16x16x32_bf16(
          false, ahi, false, blo, (short)0, acc[s], false, false);
      acc[s] = __builtin_amdgcn_wmma_f32_16x16x32_bf16(
          false, alo, false, bhi, (short)0, acc[s], false, false);
    }
  };

  // ---- pipelined main loop -------------------------------------------------
  issueA(0, 0);
  loadB(0);
  for (int ks = 0; ks < NK; ++ks) {
    const int bf = ks & 1;
    storeB(bf);
#if USE_ASYNC
    __builtin_amdgcn_s_wait_asynccnt(0);
#endif
    __syncthreads();
    if (ks + 1 < NK) {
      issueA(bf ^ 1, ks + 1);   // async copy overlaps with compute below
      loadB(ks + 1);            // global latency overlaps with compute below
    }
    compute(bf);
    __syncthreads();
  }

  // ---- epilogue: C/D layout is lane->N, VGPR r -> M = 8*half + r ----------
  float bias[8];
  #pragma unroll
  for (int r = 0; r < 8; ++r) {
    int pg = p0 + wave * 16 + half * 8 + r;
    bias[r] = (pg < P_LEN) ? (bs[pg] + bt[pg]) : 0.0f;
  }
  #pragma unroll
  for (int s = 0; s < 4; ++s) {
    int cg = c0 + s * 16 + l16;
    if (cg >= C_LEN) continue;
    #pragma unroll
    for (int r = 0; r < 8; ++r) {
      int pg = p0 + wave * 16 + half * 8 + r;
      if (pg < P_LEN) {
        float v = acc[s][r] + bias[r];
        out[((size_t)b * P_LEN + pg) * C_LEN + cg] = v > 0.0f ? v : 0.0f;
      }
    }
  }
}

// ---------------------------------------------------------------------------
extern "C" void kernel_launch(void* const* d_in, const int* in_sizes, int n_in,
                              void* d_out, int out_size, void* d_ws, size_t ws_size,
                              hipStream_t stream) {
  const float* x  = (const float*)d_in[0];   // [64, 336, 1782]
  const float* Ws = (const float*)d_in[1];   // [720, 336]
  const float* bs = (const float*)d_in[2];   // [720]
  const float* Wt = (const float*)d_in[3];   // [720, 336]
  const float* bt = (const float*)d_in[4];   // [720]
  float* out = (float*)d_out;                // [64, 720, 1782]

  __bf16* Whi = (__bf16*)d_ws;               // [768][352] bf16
  __bf16* Wlo = Whi + (size_t)PW * KP;       // [768][352] bf16

  const int prepN = PW * KP;
  prep_weights<<<(prepN + 255) / 256, 256, 0, stream>>>(Ws, Wt, Whi, Wlo);

  dim3 grid((C_LEN + TN - 1) / TN, PW / TM, 64);   // (28, 12, 64)
  dlinear_gemm<<<grid, 128, 0, stream>>>(x, Whi, Wlo, bs, bt, out);
}
